// GraphLayer_33809982554681
// MI455X (gfx1250) — matrix operand order repaired and checked by reading, not compiled, not run on hardware
//
#include <hip/hip_runtime.h>

typedef __attribute__((ext_vector_type(16))) _Float16 v16h;
typedef __attribute__((ext_vector_type(8)))  _Float16 v8h;
typedef __attribute__((ext_vector_type(8)))  float    v8f;
typedef __attribute__((ext_vector_type(4)))  float    v4f;

#define N_NODES 6
#define N_EDGES 15
#define DIM     64
#define WROW    72   // padded LDS row stride in halves: 144B -> 16B aligned, 36-dword bank stride (conflict-free)

// LDS partition sizes (bytes)
#define SW_BYTES    (N_EDGES * DIM * WROW * 2)  // 138240: f16 weights, padded rows
#define SBIAS_BYTES (5 * DIM * 4)               // 1280: per-node bias sums (f32)
#define SSTAGE_BYTES (8 * 16 * WROW * 2)        // 18432: per-wave 16-row activation staging
#define SMEM_BYTES  (SW_BYTES + SBIAS_BYTES + SSTAGE_BYTES)

__launch_bounds__(256)
__global__ void dag_gemm_kernel(const float* __restrict__ x,
                                const float* __restrict__ W,
                                const float* __restrict__ b,
                                float* __restrict__ out,
                                int batch) {
    extern __shared__ char smem[];
    _Float16* sW    = (_Float16*)smem;                       // [15*64][WROW]
    float*    sBias = (float*)(smem + SW_BYTES);             // [5][64]
    _Float16* sStg  = (_Float16*)(smem + SW_BYTES + SBIAS_BYTES); // [8][16][WROW]

    const int tid = threadIdx.x;

    // ---- Stage W (f32 global -> f16 LDS, padded rows) ----
    for (int idx = tid * 4; idx < N_EDGES * DIM * DIM; idx += 256 * 4) {
        v4f w4 = *(const v4f*)(W + idx);
        int row = idx >> 6;        // [0, 960)
        int col = idx & 63;        // multiple of 4
        _Float16* p = sW + row * WROW + col;
        p[0] = (_Float16)w4.x; p[1] = (_Float16)w4.y;
        p[2] = (_Float16)w4.z; p[3] = (_Float16)w4.w;
    }
    // ---- Per-node bias sums ----
    if (tid < DIM) {
        int e = 0;
        #pragma unroll
        for (int j = 1; j < N_NODES; ++j) {
            float s = 0.f;
            for (int i = 0; i < j; ++i, ++e) s += b[e * DIM + tid];
            sBias[(j - 1) * DIM + tid] = s;
        }
    }
    __syncthreads();

    const int wave   = tid >> 5;
    const int lane   = tid & 31;
    const int laneM  = lane & 15;   // M (A/D rows) or N (B/D cols) within tile
    const int laneHi = lane >> 4;   // selects K sub-run / upper-row half
    const long rowBase = (long)blockIdx.x * 128 + wave * 16;
    if (rowBase >= batch) return;
    _Float16* stage = sStg + wave * 16 * WROW;

    // A fragments for nodes 0..4, 2 K-chunks each (16x32 f16 per chunk)
    v16h A[5][2];

    // ---- Node 0: load x rows, convert f32 -> f16 A fragments ----
    {
        const float* xr = x + (rowBase + laneM) * DIM;
        #pragma unroll
        for (int kc = 0; kc < 2; ++kc) {
            v16h a;
            #pragma unroll
            for (int g = 0; g < 2; ++g) {
                const int k0 = kc * 32 + g * 16 + laneHi * 8;
                v4f f0 = *(const v4f*)(xr + k0);
                v4f f1 = *(const v4f*)(xr + k0 + 4);
                #pragma unroll
                for (int q = 0; q < 4; ++q) {
                    a[g * 8 + q]     = (_Float16)f0[q];
                    a[g * 8 + 4 + q] = (_Float16)f1[q];
                }
            }
            A[0][kc] = a;
        }
    }

    // ---- DAG: nodes 1..5, edge e=(i,j) index = j*(j-1)/2 + i ----
    int e = 0;
    #pragma unroll
    for (int j = 1; j < N_NODES; ++j) {
        v8f acc[4];
        #pragma unroll
        for (int nt = 0; nt < 4; ++nt) {
            const float bb = sBias[(j - 1) * DIM + nt * 16 + laneM];
            #pragma unroll
            for (int r = 0; r < 8; ++r) acc[nt][r] = bb;
        }
        #pragma unroll
        for (int i = 0; i < j; ++i, ++e) {
            #pragma unroll
            for (int kc = 0; kc < 2; ++kc) {
                #pragma unroll
                for (int nt = 0; nt < 4; ++nt) {
                    // B fragment: lane n = nt*16+laneM, K = kc*32 + laneHi*8 + [0..7], then +16
                    const _Float16* wr = sW + (e * DIM + nt * 16 + laneM) * WROW
                                            + kc * 32 + laneHi * 8;
                    v8h b0 = *(const v8h*)wr;
                    v8h b1 = *(const v8h*)(wr + 16);
                    v16h bfrag = __builtin_shufflevector(b0, b1,
                        0, 1, 2, 3, 4, 5, 6, 7, 8, 9, 10, 11, 12, 13, 14, 15);
                    acc[nt] = __builtin_amdgcn_wmma_f32_16x16x32_f16(
                        false, A[i][kc], false, bfrag, (short)0, acc[nt],
                        false, false);
                }
            }
        }
        if (j == N_NODES - 1) {
            // D layout: lane col n = nt*16+laneM, rows laneHi*8 + r
            #pragma unroll
            for (int nt = 0; nt < 4; ++nt)
                #pragma unroll
                for (int r = 0; r < 8; ++r)
                    out[(rowBase + laneHi * 8 + r) * DIM + nt * 16 + laneM] = acc[nt][r];
        } else {
            // Stage f32 D -> f16 row-major LDS tile (own-wave region, DS in-order)
            #pragma unroll
            for (int nt = 0; nt < 4; ++nt)
                #pragma unroll
                for (int r = 0; r < 8; ++r)
                    stage[(laneHi * 8 + r) * WROW + nt * 16 + laneM] =
                        (_Float16)acc[nt][r];
            // Re-read in A-fragment layout
            const _Float16* srow = stage + laneM * WROW;
            #pragma unroll
            for (int kc = 0; kc < 2; ++kc) {
                v8h h0 = *(const v8h*)(srow + kc * 32 + laneHi * 8);
                v8h h1 = *(const v8h*)(srow + kc * 32 + 16 + laneHi * 8);
                A[j][kc] = __builtin_shufflevector(h0, h1,
                    0, 1, 2, 3, 4, 5, 6, 7, 8, 9, 10, 11, 12, 13, 14, 15);
            }
        }
    }
}

extern "C" void kernel_launch(void* const* d_in, const int* in_sizes, int n_in,
                              void* d_out, int out_size, void* d_ws, size_t ws_size,
                              hipStream_t stream) {
    const float* x = (const float*)d_in[0];
    const float* W = (const float*)d_in[1];
    const float* b = (const float*)d_in[2];
    float* out = (float*)d_out;

    const int batch = in_sizes[0] / DIM;          // 262144
    const int rowsPerBlock = 128;                  // 8 waves x 16 rows
    const int grid = (batch + rowsPerBlock - 1) / rowsPerBlock;

    // Allow >64KB dynamic LDS (no-op/ignored where unsupported; not a stream op)
    (void)hipFuncSetAttribute((const void*)dag_gemm_kernel,
                              hipFuncAttributeMaxDynamicSharedMemorySize,
                              SMEM_BYTES);

    dag_gemm_kernel<<<grid, 256, SMEM_BYTES, stream>>>(x, W, b, out, batch);
}
